// AttentionLayer_850403525385
// MI455X (gfx1250) — matrix operand order repaired and checked by reading, not compiled, not run on hardware
//
#include <hip/hip_runtime.h>
#include <hip/hip_bf16.h>
#include <math.h>

// ---------------------------------------------------------------------------
// Single-query attention, B=32, S=4096, D=1024, fp32 in/out.
// Memory-bound (~0.5 FLOP/byte, 1.07 GB of K+V traffic -> ~46us floor at
// 23.3 TB/s). Split-K flash softmax: 512 blocks stream K,V exactly once.
// QK logits: v_wmma_f32_16x16x32_bf16 with K staged into per-wave LDS tiles
// via GLOBAL_LOAD_ASYNC_TO_LDS_B128 (ASYNCcnt, double-buffered, no block
// barriers in the hot loop). PV: coalesced float4 streaming (roofline-bound).
// ---------------------------------------------------------------------------

typedef __attribute__((ext_vector_type(16))) __bf16 v16bf;
typedef __attribute__((ext_vector_type(8)))  float  v8f;

#define B_       32
#define S_       4096
#define D_       1024
#define NSPLIT   16
#define SCHUNK   (S_ / NSPLIT)          // 256 keys per block
#define NTHREADS 256
#define NWAVES   (NTHREADS / 32)        // 8 waves
#define TILESPW  (SCHUNK / 16 / NWAVES) // 2 16-key tiles per wave
#define NCHUNK   (D_ / 32)              // 32 k-chunks of 32 dims
#define KROW_F32 36                     // 32 floats + pad: 16B-aligned rows,
                                        // 36*m mod 64 distinct for m=0..15

__device__ __forceinline__ unsigned int f2bf_raw(float f) {
  unsigned int u = __float_as_uint(f);
  unsigned int r = u + 0x7FFFu + ((u >> 16) & 1u); // round-to-nearest-even
  return r >> 16;
}
__device__ __forceinline__ unsigned int pack_bf2(float lo, float hi) {
  return f2bf_raw(lo) | (f2bf_raw(hi) << 16);
}

union FragAB { v16bf v; unsigned int u[8]; };

// Issue one 16x32 fp32 K tile chunk as 4 async global->LDS b128 transfers.
// Each lane moves 16B: row r = i*4 + lane/8, cols (lane&7)*4 .. +3.
// The s_wait_dscnt 0 guarantees no ds_load of the destination buffer is
// still in flight when the async DMA writes issue (DS queue vs async-LDS
// write ordering is not architecturally guaranteed); by this point the
// previous chunk's reads were already consumed, so it is effectively free.
__device__ __forceinline__ void issue_k_async(const float* kb, int row0, int d0,
                                              float* buf, int lane) {
  const int rr = lane >> 3;
  const int cc = (lane & 7) * 4;
  asm volatile("s_wait_dscnt 0x0" ::: "memory");
  #pragma unroll
  for (int i = 0; i < 4; ++i) {
    const int r = i * 4 + rr;
    const float* gsrc = kb + (size_t)(row0 + r) * D_ + d0 + cc;
    // Generic LDS pointer: low 32 bits are the LDS byte address (ISA 10.2).
    const unsigned ldsa = (unsigned)(size_t)(buf + r * KROW_F32 + cc);
    asm volatile("global_load_async_to_lds_b128 %0, %1, off"
                 :: "v"(ldsa), "v"((unsigned long long)(size_t)gsrc)
                 : "memory");
  }
}

__global__ __launch_bounds__(NTHREADS)
void attn_partial_kernel(const float* __restrict__ q,
                         const float* __restrict__ k,
                         const float* __restrict__ v,
                         const int*   __restrict__ mask,
                         float*       __restrict__ ws)
{
  const int split = blockIdx.x;
  const int b     = blockIdx.y;
  const int s0    = split * SCHUNK;
  const int tid   = threadIdx.x;
  const int wave  = tid >> 5;
  const int lane  = tid & 31;
  const int m     = lane & 15;       // row within 16x16 tile
  const int hi    = lane >> 4;       // lane half

  const float scale = 0.03125f;      // 1/sqrt(1024)

  __shared__ __align__(16) float lds_kf[NWAVES][2][16 * KROW_F32]; // 36 KB
  __shared__ unsigned int lds_qpair[D_ / 2];                       // 2 KB
  __shared__ float lds_logit[SCHUNK];                              // 1 KB
  __shared__ float lds_red[NTHREADS];                              // 1 KB

  // ---- stage q as bf16 pairs in LDS (coalesced float4 loads) ----
  {
    const float4 q4 = reinterpret_cast<const float4*>(q + (size_t)b * D_)[tid];
    lds_qpair[2 * tid + 0] = pack_bf2(q4.x, q4.y);
    lds_qpair[2 * tid + 1] = pack_bf2(q4.z, q4.w);
  }
  __syncthreads();

  const float* kb = k + ((size_t)b * S_ + s0) * (size_t)D_;

  // ---- QK^T logits via WMMA: per wave, 2 tiles of 16 keys over full D.
  // Double-buffered async global->LDS staging; no block barriers here. ----
  for (int tt = 0; tt < TILESPW; ++tt) {
    const int row0 = (wave * TILESPW + tt) * 16;  // first key of this tile
    v8f c = {};                                   // f32 accumulator

    issue_k_async(kb, row0, 0, &lds_kf[wave][0][0], lane);

    for (int ic = 0; ic < NCHUNK; ++ic) {
      const int cur = ic & 1;
      if (ic + 1 < NCHUNK) {
        issue_k_async(kb, row0, (ic + 1) * 32, &lds_kf[wave][cur ^ 1][0], lane);
        // async loads complete in order: <=4 outstanding => chunk ic resident
        asm volatile("s_wait_asynccnt 0x4" ::: "memory");
      } else {
        asm volatile("s_wait_asynccnt 0x0" ::: "memory");
      }

      const float* buf = &lds_kf[wave][cur][0];
      FragAB A, Bq;
      // A fragment (K tile 16x32 bf16, documented 16-bit A layout):
      //  lanes 0-15:  VGPR v<4 -> K=2v,2v+1 ; v>=4 -> K=16+2(v-4),..
      //  lanes 16-31: same +8
      #pragma unroll
      for (int vv = 0; vv < 8; ++vv) {
        const int k0 = ((vv < 4) ? 2 * vv : 16 + 2 * (vv - 4)) + 8 * hi;
        const float2 f = *reinterpret_cast<const float2*>(buf + m * KROW_F32 + k0);
        A.u[vv] = pack_bf2(f.x, f.y);
      }
      // B fragment (q replicated over 16 columns, 32x16 bf16):
  //  VGPR v: same K layout as A (v<4 -> K=2v,2v+1 ; v>=4 -> K=16+2(v-4),.. ; lanes 16-31 +8)
      const int d0 = ic * 32;
      #pragma unroll
      for (int vv = 0; vv < 8; ++vv)
        Bq.u[vv] = lds_qpair[d0 / 2 + 4 * hi + ((vv < 4) ? vv : 8 + (vv - 4))];

      c = __builtin_amdgcn_wmma_f32_16x16x32_bf16(
          /*neg_a=*/false, A.v, /*neg_b=*/false, Bq.v,
          /*c_mod=*/(short)0, c, /*reuse_a=*/false, /*reuse_b=*/false);
    }

    // C[m,n] identical over n (B was column-replicated q). C/D layout:
    // VGPR r: lanes 0-15 -> M=r, lanes 16-31 -> M=r+8. Take column N=0.
    if ((lane & 15) == 0) {
      const int mbase = hi * 8;
      #pragma unroll
      for (int r = 0; r < 8; ++r) {
        const int s  = row0 + mbase + r;
        const int mk = mask[b * S_ + s0 + s];
        lds_logit[s] = (mk == 0) ? -1e30f : c[r] * scale;
      }
    }
  }
  __syncthreads();

  // ---- partial softmax over this block's 256 logits ----
  const float x = lds_logit[tid];
  lds_red[tid] = x;
  __syncthreads();
  #pragma unroll
  for (int off = NTHREADS / 2; off > 0; off >>= 1) {
    if (tid < off) lds_red[tid] = fmaxf(lds_red[tid], lds_red[tid + off]);
    __syncthreads();
  }
  const float m_part = lds_red[0];
  __syncthreads();

  const float p = (m_part <= -1e29f) ? 0.0f : __expf(x - m_part);
  lds_logit[tid] = p;            // reuse as softmax weights
  lds_red[tid]   = p;
  __syncthreads();
  #pragma unroll
  for (int off = NTHREADS / 2; off > 0; off >>= 1) {
    if (tid < off) lds_red[tid] += lds_red[tid + off];
    __syncthreads();
  }
  const float l_part = lds_red[0];

  // ---- PV partial: acc[d] = sum_s w[s] * V[s,d]; pure streaming ----
  const float4* vp = reinterpret_cast<const float4*>(
      v + ((size_t)b * S_ + s0) * (size_t)D_);
  float4 acc = {0.f, 0.f, 0.f, 0.f};
  #pragma unroll 4
  for (int s = 0; s < SCHUNK; ++s) {
    __builtin_prefetch(vp + (size_t)(s + 8) * (D_ / 4) + tid, 0, 0);
    const float  w  = lds_logit[s];
    const float4 vv = vp[(size_t)s * (D_ / 4) + tid];
    acc.x += w * vv.x;
    acc.y += w * vv.y;
    acc.z += w * vv.z;
    acc.w += w * vv.w;
  }

  // ---- write partials: [B*NSPLIT*2] (m,l) then [B*NSPLIT*D] acc ----
  const int part = b * NSPLIT + split;
  reinterpret_cast<float4*>(ws + (size_t)B_ * NSPLIT * 2 +
                            (size_t)part * D_)[tid] = acc;
  if (tid == 0) {
    ws[part * 2 + 0] = m_part;
    ws[part * 2 + 1] = l_part;
  }
}

__global__ __launch_bounds__(NTHREADS)
void attn_combine_kernel(const float* __restrict__ ws,
                         float*       __restrict__ out)
{
  const int b   = blockIdx.x;
  const int tid = threadIdx.x;

  __shared__ float coef[NSPLIT];
  __shared__ float sdenom;

  if (tid == 0) {
    float M = -1e30f;
    #pragma unroll
    for (int i = 0; i < NSPLIT; ++i)
      M = fmaxf(M, ws[(b * NSPLIT + i) * 2 + 0]);
    float denom = 0.f;
    #pragma unroll
    for (int i = 0; i < NSPLIT; ++i) {
      const float mi = ws[(b * NSPLIT + i) * 2 + 0];
      const float li = ws[(b * NSPLIT + i) * 2 + 1];
      const float cc = (mi <= -1e29f) ? 0.f : __expf(mi - M);
      coef[i] = cc;
      denom  += cc * li;
    }
    sdenom = denom;
  }
  __syncthreads();

  const float inv = (sdenom > 0.f) ? (1.0f / sdenom) : 0.0f;
  const float* accb = ws + (size_t)B_ * NSPLIT * 2;
  float4 o = {0.f, 0.f, 0.f, 0.f};
  #pragma unroll
  for (int i = 0; i < NSPLIT; ++i) {
    const float4 a = reinterpret_cast<const float4*>(
        accb + (size_t)(b * NSPLIT + i) * D_)[tid];
    const float cc = coef[i];
    o.x += cc * a.x; o.y += cc * a.y; o.z += cc * a.z; o.w += cc * a.w;
  }
  o.x *= inv; o.y *= inv; o.z *= inv; o.w *= inv;
  reinterpret_cast<float4*>(out + (size_t)b * D_)[tid] = o;
}

extern "C" void kernel_launch(void* const* d_in, const int* in_sizes, int n_in,
                              void* d_out, int out_size, void* d_ws, size_t ws_size,
                              hipStream_t stream) {
  const float* q    = (const float*)d_in[0];   // [B, D]
  const float* k    = (const float*)d_in[1];   // [B, S, D]
  const float* v    = (const float*)d_in[2];   // [B, S, D]
  const int*   mask = (const int*)  d_in[3];   // [B, S]
  float* out = (float*)d_out;                  // [B, 1, D]
  float* ws  = (float*)d_ws;                   // needs (1024 + 512*1024)*4 ~ 2.1 MB

  dim3 grid1(NSPLIT, B_);
  attn_partial_kernel<<<grid1, NTHREADS, 0, stream>>>(q, k, v, mask, ws);
  attn_combine_kernel<<<B_, NTHREADS, 0, stream>>>(ws, out);
}
